// ManualLSTM_46892452937740
// MI455X (gfx1250) — matrix operand order repaired and checked by reading, not compiled, not run on hardware
//
#include <hip/hip_runtime.h>
#include <hip/hip_bf16.h>

// ---------------- problem constants ----------------
#define HID    1024
#define FOURH  4096
#define BATCH  32
#define SEQ    512
#define BS     (BATCH * SEQ)     // 16384 rows of x
#define KT_N   (HID / 32)        // 32 K-tiles of 32

// ---------------- WMMA types ----------------
typedef __attribute__((ext_vector_type(16))) __bf16 v16bf;
typedef __attribute__((ext_vector_type(8)))  float  v8f;

union Frag {
    v16bf v;
    uint4 q[2];
};

__device__ inline v8f wmma_bf16(const Frag& a, const Frag& b, v8f c) {
    return __builtin_amdgcn_wmma_f32_16x16x32_bf16(
        /*neg_a=*/false, a.v, /*neg_b=*/false, b.v,
        /*c_mod=*/(short)0, c, /*reuse_a=*/false, /*reuse_b=*/false);
}

__device__ inline v8f zero8() {
    v8f z = {0.f, 0.f, 0.f, 0.f, 0.f, 0.f, 0.f, 0.f};
    return z;
}

// A fragment (16x32 bf16, row-major source with leading dim ld).
// ISA layout: lane<16 -> row m0+lane, K = {k0..k0+7, k0+16..k0+23}
//             lane>=16 -> row m0+lane-16, K = {k0+8..k0+15, k0+24..k0+31}
__device__ inline Frag load_a(const unsigned short* __restrict__ base, int ld,
                              int m0, int k0, int lane) {
    int row = m0 + (lane & 15);
    int ko  = k0 + ((lane & 16) ? 8 : 0);
    const uint4* p = reinterpret_cast<const uint4*>(base + (size_t)row * ld + ko);
    Frag f;
    f.q[0] = p[0];   // 8 bf16: K = ko .. ko+7
    f.q[1] = p[2];   // 8 bf16: K = ko+16 .. ko+23
    return f;
}

// Packed B layout, planar per (nt,kt) block:
//   block = 512 bf16: plane0 = 32 lanes x 8 bf16 (v elems 0..7),
//                     plane1 = 32 lanes x 8 bf16 (v elems 8..15).
// Lane stride inside a plane = 16 B -> conflict-free ds_load_b128,
// and two coalesced 512 B segments for global loads.
__device__ inline Frag load_b(const unsigned short* __restrict__ packed,
                              int nt, int kt, int lane) {
    const uint4* p = reinterpret_cast<const uint4*>(
        packed + (size_t)(nt * KT_N + kt) * 512);
    Frag f;
    f.q[0] = p[lane];
    f.q[1] = p[32 + lane];
    return f;
}

__device__ inline Frag load_b_lds(const unsigned short* sh,
                                  int nt_local, int kt, int lane) {
    const uint4* p = reinterpret_cast<const uint4*>(
        sh + (size_t)(nt_local * KT_N + kt) * 512);
    Frag f;
    f.q[0] = p[lane];
    f.q[1] = p[32 + lane];
    return f;
}

__device__ inline unsigned short f32_to_bf16(float f) {
    unsigned int u = __float_as_uint(f);
    unsigned int r = u + 0x7FFFu + ((u >> 16) & 1u);   // round-to-nearest-even
    return (unsigned short)(r >> 16);
}

__device__ inline float sigm(float x) { return 1.0f / (1.0f + __expf(-x)); }
__device__ inline float tanh_fast(float x) {
    x = fminf(fmaxf(x, -15.f), 15.f);
    float e = __expf(2.f * x);
    return (e - 1.f) / (e + 1.f);
}

// ---------------- grid-wide sense barrier ----------------
__device__ inline void grid_barrier(unsigned int* cnt, unsigned int* gen,
                                    unsigned int nblocks) {
    __syncthreads();
    if (threadIdx.x == 0) {
        __threadfence();
        unsigned int g = __hip_atomic_load(gen, __ATOMIC_RELAXED,
                                           __HIP_MEMORY_SCOPE_AGENT);
        unsigned int a = __hip_atomic_fetch_add(cnt, 1u, __ATOMIC_ACQ_REL,
                                                __HIP_MEMORY_SCOPE_AGENT);
        if (a == nblocks - 1u) {
            __hip_atomic_store(cnt, 0u, __ATOMIC_RELAXED,
                               __HIP_MEMORY_SCOPE_AGENT);
            __hip_atomic_fetch_add(gen, 1u, __ATOMIC_RELEASE,
                                   __HIP_MEMORY_SCOPE_AGENT);
        } else {
            while (__hip_atomic_load(gen, __ATOMIC_ACQUIRE,
                                     __HIP_MEMORY_SCOPE_AGENT) == g) {
                __builtin_amdgcn_s_sleep(2);
            }
        }
        __threadfence();
    }
    __syncthreads();
}

// ---------------- prep kernels ----------------
__global__ void lstm_init_sync(unsigned int* bar) {
    bar[0] = 0u;
    bar[1] = 0u;
}

// pack fp32 [K=HID, N=FOURH] row-major -> planar WMMA-B bf16 blocks
__global__ void lstm_pack_b(const float* __restrict__ w,
                            unsigned short* __restrict__ dst) {
    size_t i = (size_t)blockIdx.x * 256 + threadIdx.x;  // < HID*FOURH
    int e  = (int)(i & 7);
    int l  = (int)((i >> 3) & 31);
    int pl = (int)((i >> 8) & 1);
    int kt = (int)((i >> 9) & 31);
    int nt = (int)(i >> 14);
    int n = (nt << 4) + (l & 15);
    int k = (kt << 5) + ((l & 16) ? 8 : 0) + (pl << 4) + e;
    dst[i] = f32_to_bf16(w[(size_t)k * FOURH + n]);
}

__global__ void lstm_cvt_bf16(const float* __restrict__ src,
                              unsigned short* __restrict__ dst) {
    size_t i = (size_t)blockIdx.x * 256 + threadIdx.x;
    dst[i] = f32_to_bf16(src[i]);
}

// ---------------- big input GEMM: xg = x @ W_in + bias ----------------
// grid (256, 32), block 256 (8 waves). Block tile: 64 rows x 128 cols.
// Each wave: 32x32 output (2 M-tiles x 2 N-tiles).
__global__ void __launch_bounds__(256)
lstm_input_gemm(const unsigned short* __restrict__ xb,
                const unsigned short* __restrict__ wip,
                const float* __restrict__ bias,
                float* __restrict__ xg) {
    const int lane = threadIdx.x & 31;
    const int wave = threadIdx.x >> 5;
    const int wm = wave >> 2, wn = wave & 3;
    const int m_base = blockIdx.x * 64 + wm * 32;
    const int n_base = blockIdx.y * 128 + wn * 32;
    const int nt0 = n_base >> 4;

    v8f acc00 = zero8(), acc01 = zero8(), acc10 = zero8(), acc11 = zero8();

    for (int kt = 0; kt < KT_N; ++kt) {
        const int k0 = kt * 32;
        // prefetch next K-tile of the streaming x operand
        __builtin_prefetch(xb + (size_t)(m_base + (lane & 15)) * HID + k0 + 32, 0, 0);
        Frag A0 = load_a(xb, HID, m_base,      k0, lane);
        Frag A1 = load_a(xb, HID, m_base + 16, k0, lane);
        Frag B0 = load_b(wip, nt0,     kt, lane);
        Frag B1 = load_b(wip, nt0 + 1, kt, lane);
        acc00 = wmma_bf16(A0, B0, acc00);
        acc01 = wmma_bf16(A0, B1, acc01);
        acc10 = wmma_bf16(A1, B0, acc10);
        acc11 = wmma_bf16(A1, B1, acc11);
    }

    const int hi8 = (lane & 16) ? 8 : 0;
    const int l15 = lane & 15;
    const int col0 = n_base + l15;
    const int col1 = col0 + 16;
    const float bia0 = bias[col0];
    const float bia1 = bias[col1];
#pragma unroll
    for (int r = 0; r < 8; ++r) {
        int row0 = m_base + r + hi8;
        int row1 = row0 + 16;
        xg[(size_t)row0 * FOURH + col0] = acc00[r] + bia0;
        xg[(size_t)row0 * FOURH + col1] = acc01[r] + bia1;
        xg[(size_t)row1 * FOURH + col0] = acc10[r] + bia0;
        xg[(size_t)row1 * FOURH + col1] = acc11[r] + bia1;
    }
}

// ---------------- persistent recurrent kernel ----------------
// 32 blocks x 256 threads = 256 waves; wave W owns N-columns [16W, 16W+16).
// Block's 256 KB W_h slice is staged in LDS once (reused 512x).
// Per step: gates = h @ W_h + xg_t  (WMMA), barrier, pointwise update, barrier.
__global__ void __launch_bounds__(256)
lstm_recurrent(const float* __restrict__ xg,
               const unsigned short* __restrict__ whp,
               unsigned short* __restrict__ hb,   // h in bf16 [BATCH, HID]
               float* __restrict__ c,             // cell state [BATCH, HID]
               float* __restrict__ gates,         // [BATCH, FOURH]
               unsigned int* __restrict__ bar,
               float* __restrict__ out)           // [BATCH, SEQ, HID]
{
    __shared__ __align__(16) unsigned short wh_s[8 * KT_N * 512];  // 256 KB

    const int lane = threadIdx.x & 31;
    const int wave = threadIdx.x >> 5;
    const int W    = blockIdx.x * 8 + wave;   // 0..255
    const int n0   = W * 16;
    const int gtid = blockIdx.x * 256 + threadIdx.x;   // 0..8191
    const int hi8  = (lane & 16) ? 8 : 0;
    const int l15  = lane & 15;
    const int col  = n0 + l15;

    // stage this block's W_h slice into LDS (once, reused all 512 steps)
    {
        const uint4* src = reinterpret_cast<const uint4*>(
            whp + (size_t)blockIdx.x * 8 * KT_N * 512);
        uint4* dst = reinterpret_cast<uint4*>(wh_s);
        for (int i = threadIdx.x; i < (8 * KT_N * 512) / 8; i += 256)
            dst[i] = src[i];
    }

    // zero-init h, c
    for (int cell = gtid; cell < BATCH * HID; cell += 32 * 256) {
        c[cell] = 0.f;
        hb[cell] = 0;
    }
    grid_barrier(bar, bar + 1, 32);   // also covers the LDS staging

    for (int t = 0; t < SEQ; ++t) {
        // ---- GEMM phase: 32 x 16 tile of h @ W_h ----
        v8f acc0 = zero8();   // batch rows 0..15
        v8f acc1 = zero8();   // batch rows 16..31
        for (int kt = 0; kt < KT_N; ++kt) {
            const int k0 = kt * 32;
            Frag A0 = load_a(hb, HID, 0,  k0, lane);
            Frag A1 = load_a(hb, HID, 16, k0, lane);
            Frag Bf = load_b_lds(wh_s, wave, kt, lane);
            acc0 = wmma_bf16(A0, Bf, acc0);
            acc1 = wmma_bf16(A1, Bf, acc1);
        }
#pragma unroll
        for (int r = 0; r < 8; ++r) {
            int b0 = r + hi8;        // batch row of acc0 element r
            int b1 = b0 + 16;        // batch row of acc1 element r
            gates[(size_t)b0 * FOURH + col] =
                acc0[r] + xg[((size_t)b0 * SEQ + t) * FOURH + col];
            gates[(size_t)b1 * FOURH + col] =
                acc1[r] + xg[((size_t)b1 * SEQ + t) * FOURH + col];
        }
        grid_barrier(bar, bar + 1, 32);

        // ---- pointwise LSTM cell update ----
        for (int cell = gtid; cell < BATCH * HID; cell += 32 * 256) {
            int b = cell >> 10;
            int j = cell & (HID - 1);
            const float* gb = gates + (size_t)b * FOURH;
            float ig = gb[j];
            float fg = gb[HID + j];
            float gg = gb[2 * HID + j];
            float og = gb[3 * HID + j];
            float cn = sigm(fg) * c[cell] + sigm(ig) * tanh_fast(gg);
            float hn = sigm(og) * tanh_fast(cn);
            c[cell] = cn;
            out[((size_t)b * SEQ + t) * HID + j] = hn;
            hb[cell] = f32_to_bf16(hn);
        }
        grid_barrier(bar, bar + 1, 32);
    }
}

// ---------------- host launch ----------------
extern "C" void kernel_launch(void* const* d_in, const int* in_sizes, int n_in,
                              void* d_out, int out_size, void* d_ws, size_t ws_size,
                              hipStream_t stream) {
    (void)in_sizes; (void)n_in; (void)out_size; (void)ws_size;

    const float* x    = (const float*)d_in[0];
    const float* w_in = (const float*)d_in[1];
    const float* w_h  = (const float*)d_in[2];
    const float* bias = (const float*)d_in[3];
    float* out = (float*)d_out;

    char* ws = (char*)d_ws;
    size_t off = 0;
    float* xg = (float*)(ws + off);               off += (size_t)BS * FOURH * 4;   // 256 MiB
    unsigned short* xb   = (unsigned short*)(ws + off); off += (size_t)BS * HID * 2;     // 32 MiB
    unsigned short* wi_p = (unsigned short*)(ws + off); off += (size_t)HID * FOURH * 2;  // 8 MiB
    unsigned short* wh_p = (unsigned short*)(ws + off); off += (size_t)HID * FOURH * 2;  // 8 MiB
    unsigned short* hb   = (unsigned short*)(ws + off); off += (size_t)BATCH * HID * 2;
    float* c     = (float*)(ws + off);            off += (size_t)BATCH * HID * 4;
    float* gates = (float*)(ws + off);            off += (size_t)BATCH * FOURH * 4;
    unsigned int* bar = (unsigned int*)(ws + off); off += 2 * sizeof(unsigned int);

    lstm_init_sync<<<1, 1, 0, stream>>>(bar);

    // pack weights to planar bf16 WMMA-B layout
    lstm_pack_b<<<(HID * FOURH) / 256, 256, 0, stream>>>(w_in, wi_p);
    lstm_pack_b<<<(HID * FOURH) / 256, 256, 0, stream>>>(w_h, wh_p);

    // x -> bf16 row-major
    lstm_cvt_bf16<<<((size_t)BS * HID) / 256, 256, 0, stream>>>(x, xb);

    // xg = x @ W_in + bias
    lstm_input_gemm<<<dim3(BS / 64, FOURH / 128), 256, 0, stream>>>(xb, wi_p, bias, xg);

    // persistent recurrence over all 512 timesteps
    lstm_recurrent<<<32, 256, 0, stream>>>(xg, wh_p, hb, c, gates, bar, out);
}
